// CreatorwyxGroupedDINAdapter_32779190403193
// MI455X (gfx1250) — matrix operand order repaired and checked by reading, not compiled, not run on hardware
//
#include <hip/hip_runtime.h>
#include <stdint.h>

#define DEV __device__ __forceinline__

typedef __attribute__((ext_vector_type(16))) __bf16 v16bf;
typedef __attribute__((ext_vector_type(8)))  float  v8f;
typedef __attribute__((ext_vector_type(4)))  int    v4i;

static constexpr int B_ = 2048, SC_ = 20, SX_ = 20, S_ = 200;
static constexpr int D_ = 64, H_ = 256, DENSE_ = 128, NH_ = 3;
static constexpr int STILE = 16, NTILES = (S_ + STILE - 1) / STILE; // 13

#if defined(__has_builtin)
#if __has_builtin(__builtin_amdgcn_global_load_async_to_lds_b128) && \
    __has_builtin(__builtin_amdgcn_s_wait_asynccnt)
#define HAVE_ASYNC_LDS 1
#endif
#endif

// ---------- bf16 helpers (native __bf16 converts) ----------
DEV unsigned short f2us(float f) {
  __bf16 b = (__bf16)f;
  unsigned short u;
  __builtin_memcpy(&u, &b, 2);
  return u;
}
DEV float us2f(unsigned short u) { return __uint_as_float(((unsigned)u) << 16); }
// pack two floats into one dword of bf16 pair {hi,lo}
DEV unsigned packbf(float lo, float hi) {
  union { unsigned u; __bf16 b[2]; } x;
  x.b[0] = (__bf16)lo;
  x.b[1] = (__bf16)hi;
  return x.u;
}
DEV float siluf(float x) { return x * (1.f / (1.f + __expf(-x))); }

// ---------- WMMA fragment helpers (CDNA5 ISA layouts, wave32) ----------
// A (16x32 bf16): lane<16 -> M=lane, e<8 => K=e, e>=8 => K=e+8
//                 lane>=16 -> M=lane-16, e<8 => K=e+8, e>=8 => K=e+16
DEV int kidxA(int lane, int e) {
  if (lane < 16) return (e < 8) ? e : (e + 8);
  return (e < 8) ? (e + 8) : (e + 16);
}
// A-frag dword j holds K pair {k0,k0+1}, k0 = kidxA(lane,2j):
DEV int wIdxA(int lane, int j) { return j + ((j >= 4) ? 4 : 0) + ((lane & 16) ? 4 : 0); }

DEV v16bf loadAfragLDS(const unsigned short* row16, int ks, int lane) {
  const unsigned* row = (const unsigned*)row16;
  union { v16bf v; unsigned w[8]; } u;
#pragma unroll
  for (int j = 0; j < 8; ++j) u.w[j] = row[ks * 16 + wIdxA(lane, j)];
  return u.v;
}
// B fragments pre-swizzled by k_prep: frag fi=ks*4+nt, 8 dwords per lane
DEV v16bf loadBfragSw(const unsigned* Wsw, int fi, int lane) {
  union { v16bf v; uint4 q[2]; } u;
  const uint4* p = (const uint4*)(Wsw + (size_t)(fi * 32 + lane) * 8);
  u.q[0] = p[0];
  u.q[1] = p[1];
  return u.v;
}
DEV v8f wmma_bf16(v16bf a, v16bf b, v8f c) {
  return __builtin_amdgcn_wmma_f32_16x16x32_bf16(false, a, false, b, (short)0, c,
                                                 false, false);
}

// decompose swizzled word index -> (k0, n)
DEV void widx_decode(int w, int& k0, int& n) {
  int j = w & 7, lane = (w >> 3) & 31, fi = w >> 8;
  int nt = fi & 3, ks = fi >> 2;
  n = nt * 16 + (lane & 15);
  k0 = ks * 32 + ((lane & 16) ? 8 : 0) + ((j >= 4) ? 8 : 0) + 2 * j;
}

// ======================================================================
// Kernel 0: weight prep -> pre-swizzled bf16 fragment buffers
// ======================================================================
__global__ void k_prep(const float* tw2, const float* cw, const float* w1_a,
                       const float* w1_c, const float* w1_i, unsigned* o_tw2,
                       unsigned* o_cw, unsigned* o_wq, unsigned* o_wk,
                       unsigned* o_wc) {
  int i = blockIdx.x * blockDim.x + threadIdx.x;
  if (i < 2048) {
    int k0, n;
    widx_decode(i, k0, n);
    o_tw2[i] = packbf(tw2[k0 * 64 + n], tw2[(k0 + 1) * 64 + n]);
  }
  if (i < 4096) {
    int k0, n;
    widx_decode(i, k0, n);
    o_cw[i] = packbf(cw[k0 * 64 + n], cw[(k0 + 1) * 64 + n]);
  }
  if (i < 2048) {
    int k0, n;
    widx_decode(i, k0, n);
    const float* Ws[3] = {w1_a, w1_c, w1_i};
#pragma unroll
    for (int h = 0; h < 3; ++h) {
      const float* W = Ws[h];
      float a0 = W[k0 * 64 + n], b0 = W[(64 + k0) * 64 + n];
      float c0 = W[(128 + k0) * 64 + n], d0 = W[(192 + k0) * 64 + n];
      float a1 = W[(k0 + 1) * 64 + n], b1 = W[(64 + k0 + 1) * 64 + n];
      float c1 = W[(128 + k0 + 1) * 64 + n], d1 = W[(192 + k0 + 1) * 64 + n];
      o_wq[h * 2048 + i] = packbf(a0 + d0, a1 + d1);
      o_wk[h * 2048 + i] = packbf(b0 - d0, b1 - d1);
      o_wc[h * 2048 + i] = packbf(c0, c1);
    }
  }
}

// ======================================================================
// Kernel 1: candidate / context masked-mean summaries
// ======================================================================
__global__ void __launch_bounds__(64) k_summ(const int* ctok,
                                             const unsigned char* cmask,
                                             const int* xtok,
                                             const unsigned char* xmask,
                                             const float* tok, float* cand_sum,
                                             float* ctx_sum) {
  int b = blockIdx.x, d = threadIdx.x;
  float acc = 0.f, cnt = 0.f;
  for (int i = 0; i < SC_; ++i)
    if (cmask[b * SC_ + i]) { acc += tok[(size_t)ctok[b * SC_ + i] * D_ + d]; cnt += 1.f; }
  cand_sum[b * D_ + d] = acc / fmaxf(cnt, 1.f);
  acc = 0.f; cnt = 0.f;
  for (int i = 0; i < SX_; ++i)
    if (xmask[b * SX_ + i]) { acc += tok[(size_t)xtok[b * SX_ + i] * D_ + d]; cnt += 1.f; }
  ctx_sum[b * D_ + d] = acc / fmaxf(cnt, 1.f);
}

// ======================================================================
// Kernel 2: history pipeline: time MLP + comp GEMM + LN + SiLU (WMMA)
// ======================================================================
__global__ void __launch_bounds__(32) k_hist(
    const float* htime, const int* htok, const int* hpos, const int* hgrp,
    const unsigned char* hmask, const float* tok, const float* pos,
    const float* grp, const float* tw1, const float* tb1, const float* tb2,
    const float* comp_b, const float* comp_g, const float* comp_be,
    const unsigned* tw2_sw, const unsigned* cw_sw, unsigned short* hist_emb,
    float* comp_sum, float* mask_cnt) {
  __shared__ unsigned sPw[16][32];        // pos+grp packed bf16
  __shared__ unsigned short sE16[16][64]; // time_e packed bf16
  __shared__ float sC[16][64];
  __shared__ float v_tw1[64], v_tb1[64], v_tb2[64], v_cb[64], v_cg[64], v_cbe[64];
  __shared__ float sm[16], sr[16];
  __shared__ float sRed[32];
  __shared__ int sPi[16], sGi[16], sTi[16], sMk[16];

  int bx = blockIdx.x;
  int b = bx / NTILES, tile = bx % NTILES;
  int lane = threadIdx.x;
  int s0 = tile * STILE;

  for (int i = lane; i < 64; i += 32) {
    v_tw1[i] = tw1[i]; v_tb1[i] = tb1[i]; v_tb2[i] = tb2[i];
    v_cb[i] = comp_b[i]; v_cg[i] = comp_g[i]; v_cbe[i] = comp_be[i];
  }
  if (lane < 16) {
    int s = s0 + lane;
    bool v = (s < S_);
    sPi[lane] = v ? hpos[b * S_ + s] : 0;
    sGi[lane] = v ? hgrp[b * S_ + s] : 0;
    sTi[lane] = v ? htok[b * S_ + s] : 0;
    sMk[lane] = (v && hmask[b * S_ + s]) ? 1 : 0;
  }
  __syncthreads();

  for (int i = lane; i < 16 * 32; i += 32) {
    int r = i >> 5, w = i & 31, s = s0 + r;
    unsigned val = 0u;
    if (s < S_) {
      float2 p2 = ((const float2*)(pos + (size_t)sPi[r] * D_))[w];
      float2 g2 = ((const float2*)(grp + (size_t)sGi[r] * D_))[w];
      val = packbf(p2.x + g2.x, p2.y + g2.y);
    }
    sPw[r][w] = val;
  }

  int row = lane & 15;
  int ncol = lane & 15;
  int s_row = s0 + row;
  float t = (s_row < S_) ? htime[b * S_ + s_row] : 0.f;

  // time activation A-fragments (K=64 -> 2 k-steps), packed pairs
  v16bf tA[2];
#pragma unroll
  for (int ks = 0; ks < 2; ++ks) {
    union { v16bf v; unsigned w[8]; } u;
#pragma unroll
    for (int j = 0; j < 8; ++j) {
      int k0 = ks * 32 + kidxA(lane, 2 * j);
      float x0 = siluf(t * v_tw1[k0] + v_tb1[k0]);
      float x1 = siluf(t * v_tw1[k0 + 1] + v_tb1[k0 + 1]);
      u.w[j] = packbf(x0, x1);
    }
    tA[ks] = u.v;
  }
  // time_e = act @ time_w2 + b2 -> sE16 (bf16)
#pragma unroll
  for (int nt = 0; nt < 4; ++nt) {
    v8f c;
    float ci = v_tb2[nt * 16 + ncol];
#pragma unroll
    for (int r = 0; r < 8; ++r) c[r] = ci;
#pragma unroll
    for (int ks = 0; ks < 2; ++ks)
      c = wmma_bf16(tA[ks], loadBfragSw(tw2_sw, ks * 4 + nt, lane), c);
    int rb = (lane >> 4) * 8;
#pragma unroll
    for (int r = 0; r < 8; ++r) sE16[rb + r][nt * 16 + ncol] = f2us(c[r]);
  }
  __syncthreads();

  // comp = [pos+grp | time_e] @ comp_w + comp_b  (K=128 -> 4 k-steps)
  v16bf cA[4];
  cA[0] = loadAfragLDS((const unsigned short*)&sPw[row][0], 0, lane);
  cA[1] = loadAfragLDS((const unsigned short*)&sPw[row][0], 1, lane);
  cA[2] = loadAfragLDS(&sE16[row][0], 0, lane);
  cA[3] = loadAfragLDS(&sE16[row][0], 1, lane);
#pragma unroll
  for (int nt = 0; nt < 4; ++nt) {
    v8f c;
    float ci = v_cb[nt * 16 + ncol];
#pragma unroll
    for (int r = 0; r < 8; ++r) c[r] = ci;
#pragma unroll
    for (int ks = 0; ks < 4; ++ks)
      c = wmma_bf16(cA[ks], loadBfragSw(cw_sw, ks * 4 + nt, lane), c);
    int rb = (lane >> 4) * 8;
#pragma unroll
    for (int r = 0; r < 8; ++r) sC[rb + r][nt * 16 + ncol] = c[r];
  }
  __syncthreads();

  // LN stats: split 64-col sums across both half-waves
  {
    int r = lane & 15, c0 = (lane & 16) ? 32 : 0;
    float p = 0.f;
    for (int c = 0; c < 32; ++c) p += sC[r][c0 + c];
    sRed[lane] = p;
  }
  __syncthreads();
  if (lane < 16) sm[lane] = (sRed[lane] + sRed[lane + 16]) * (1.f / 64.f);
  __syncthreads();
  {
    int r = lane & 15, c0 = (lane & 16) ? 32 : 0;
    float m = sm[r], p = 0.f;
    for (int c = 0; c < 32; ++c) { float d0 = sC[r][c0 + c] - m; p += d0 * d0; }
    sRed[lane] = p;
  }
  __syncthreads();
  if (lane < 16)
    sr[lane] = rsqrtf((sRed[lane] + sRed[lane + 16]) * (1.f / 64.f) + 1e-5f);
  __syncthreads();

  // epilogue: each lane owns column pair {2*lane, 2*lane+1}
  float2 cg = ((const float2*)v_cg)[lane];
  float2 cb = ((const float2*)v_cbe)[lane];
  float acc0 = 0.f, acc1 = 0.f;
  unsigned* ghe = (unsigned*)hist_emb;
  for (int r = 0; r < 16; ++r) {
    int s = s0 + r;
    float m = sm[r], rs = sr[r];
    float2 cv = ((const float2*)&sC[r][0])[lane];
    float comp0 = siluf((cv.x - m) * rs * cg.x + cb.x);
    float comp1 = siluf((cv.y - m) * rs * cg.y + cb.y);
    if (sMk[r]) { acc0 += comp0; acc1 += comp1; }
    if (s < S_) {
      float2 tk = ((const float2*)(tok + (size_t)sTi[r] * D_))[lane];
      ghe[(size_t)(b * S_ + s) * 32 + lane] = packbf(tk.x + comp0, tk.y + comp1);
    }
  }
  atomicAdd(&comp_sum[b * D_ + 2 * lane], acc0);
  atomicAdd(&comp_sum[b * D_ + 2 * lane + 1], acc1);
  if (lane == 0) {
    int nm = 0;
    for (int r = 0; r < 16; ++r) nm += sMk[r];
    atomicAdd(&mask_cnt[b], (float)nm);
  }
}

// ======================================================================
// Kernel 3: qW[b,h,:] = cand_summary @ (W1a+W1d)_h + b1_h   (WMMA)
// ======================================================================
__global__ void __launch_bounds__(32) k_qw(const float* cand_sum,
                                           const unsigned* wq_sw,
                                           const float* b1_0, const float* b1_1,
                                           const float* b1_2, float* qw) {
  int b0 = blockIdx.x * 16;
  int lane = threadIdx.x;
  int row = lane & 15, ncol = lane & 15;
  const float2* arow = (const float2*)(cand_sum + (size_t)(b0 + row) * D_);
  v16bf A[2];
#pragma unroll
  for (int ks = 0; ks < 2; ++ks) {
    union { v16bf v; unsigned w[8]; } u;
#pragma unroll
    for (int j = 0; j < 8; ++j) {
      int k0 = ks * 32 + kidxA(lane, 2 * j);
      float2 v2 = arow[k0 >> 1];
      u.w[j] = packbf(v2.x, v2.y);
    }
    A[ks] = u.v;
  }
  const float* b1s[3] = {b1_0, b1_1, b1_2};
  for (int h = 0; h < 3; ++h) {
#pragma unroll
    for (int nt = 0; nt < 4; ++nt) {
      v8f c;
      float ci = b1s[h][nt * 16 + ncol];
#pragma unroll
      for (int r = 0; r < 8; ++r) c[r] = ci;
#pragma unroll
      for (int ks = 0; ks < 2; ++ks)
        c = wmma_bf16(A[ks], loadBfragSw(wq_sw + h * 2048, ks * 4 + nt, lane), c);
      int rb = (lane >> 4) * 8;
#pragma unroll
      for (int r = 0; r < 8; ++r)
        qw[(size_t)(b0 + rb + r) * (NH_ * D_) + h * D_ + nt * 16 + ncol] = c[r];
    }
  }
}

// ======================================================================
// Kernel 4: DIN activation unit (WMMA) -> masked logits per head
// ======================================================================
__global__ void __launch_bounds__(32) k_din(
    const unsigned short* hist_emb, const float* cand_sum, const float* qw,
    const unsigned* wk_sw, const unsigned* wc_sw, const float* w2_0,
    const float* w2_1, const float* w2_2, const float* b2_0, const float* b2_1,
    const float* b2_2, const int* hgrp, const unsigned char* hmask,
    float* logits) {
  __shared__ unsigned sKw[16][32]; // keys packed bf16
  __shared__ float sq[64];
  __shared__ float sH[16][64];
  __shared__ float sw2[3][64];
  __shared__ float sRed[32];
  int bx = blockIdx.x;
  int b = bx / NTILES, tile = bx % NTILES;
  int lane = threadIdx.x;
  int s0 = tile * STILE;

  for (int i = lane; i < 64; i += 32) {
    sq[i] = cand_sum[b * D_ + i];
    sw2[0][i] = w2_0[i]; sw2[1][i] = w2_1[i]; sw2[2][i] = w2_2[i];
  }
  const unsigned* gkeys = (const unsigned*)hist_emb;
#ifdef HAVE_ASYNC_LDS
  for (int i = lane; i < 16 * 8; i += 32) { // 8 b128 chunks per row, 128 per 16-row tile
    int r = i >> 3, q = i & 7, s = s0 + r;
    if (s < S_) {
      v4i* src = (v4i*)(gkeys + (size_t)(b * S_ + s) * 32 + q * 4);
      v4i* dst = (v4i*)&sKw[r][q * 4];
      __builtin_amdgcn_global_load_async_to_lds_b128(src, dst, 0, 0);
    } else {
      sKw[r][q * 4 + 0] = 0u; sKw[r][q * 4 + 1] = 0u;
      sKw[r][q * 4 + 2] = 0u; sKw[r][q * 4 + 3] = 0u;
    }
  }
  __builtin_amdgcn_s_wait_asynccnt(0);
#else
  for (int i = lane; i < 16 * 32; i += 32) {
    int r = i >> 5, w = i & 31, s = s0 + r;
    sKw[r][w] = (s < S_) ? gkeys[(size_t)(b * S_ + s) * 32 + w] : 0u;
  }
#endif
  __syncthreads();

  int row = lane & 15, ncol = lane & 15;
  // kA: raw packed keys, no conversion. kB: q (*) k, repacked.
  v16bf kA[2], kB[2];
#pragma unroll
  for (int ks = 0; ks < 2; ++ks) {
    kA[ks] = loadAfragLDS((const unsigned short*)&sKw[row][0], ks, lane);
    union { v16bf v; unsigned w[8]; } u;
#pragma unroll
    for (int j = 0; j < 8; ++j) {
      unsigned kw = sKw[row][ks * 16 + wIdxA(lane, j)];
      int k0 = ks * 32 + kidxA(lane, 2 * j);
      float lo = __uint_as_float(kw << 16);
      float hi = __uint_as_float(kw & 0xffff0000u);
      u.w[j] = packbf(sq[k0] * lo, sq[k0 + 1] * hi);
    }
    kB[ks] = u.v;
  }
  float b2v[3] = {b2_0[0], b2_1[0], b2_2[0]};

  for (int h = 0; h < 3; ++h) {
#pragma unroll
    for (int nt = 0; nt < 4; ++nt) {
      v8f c;
      float ci = qw[(size_t)b * (NH_ * D_) + h * D_ + nt * 16 + ncol];
#pragma unroll
      for (int r = 0; r < 8; ++r) c[r] = ci;
#pragma unroll
      for (int ks = 0; ks < 2; ++ks) {
        c = wmma_bf16(kA[ks], loadBfragSw(wk_sw + h * 2048, ks * 4 + nt, lane), c);
        c = wmma_bf16(kB[ks], loadBfragSw(wc_sw + h * 2048, ks * 4 + nt, lane), c);
      }
      int rb = (lane >> 4) * 8;
#pragma unroll
      for (int r = 0; r < 8; ++r) sH[rb + r][nt * 16 + ncol] = c[r];
    }
    __syncthreads();
    // logit dot split across both half-waves
    {
      int r2 = lane & 15, c0 = (lane & 16) ? 32 : 0;
      float p = 0.f;
      for (int k = 0; k < 32; ++k) {
        int c = c0 + k;
        p += siluf(sH[r2][c]) * sw2[h][c];
      }
      sRed[lane] = p;
    }
    __syncthreads();
    if (lane < 16) {
      float acc = b2v[h] + sRed[lane] + sRed[lane + 16];
      int s = s0 + lane;
      if (s < S_) {
        bool msk = (hmask[b * S_ + s] != 0) && (hgrp[b * S_ + s] == h + 1);
        logits[((size_t)b * NH_ + h) * S_ + s] = msk ? acc : -1e9f;
      }
    }
    __syncthreads();
  }
}

// ======================================================================
// Kernel 5: per-(b,head) softmax over S + weighted key sum
// ======================================================================
__global__ void __launch_bounds__(64) k_att(const float* logits,
                                            const unsigned short* hist_emb,
                                            float* head_sum) {
  __shared__ float sK[S_][64];
  __shared__ float sp[S_];
  __shared__ float red[64];
  __shared__ float sstat[2];
  int b = blockIdx.x, t = threadIdx.x;
  const unsigned* gkeys = (const unsigned*)hist_emb;
  for (int i = t; i < S_ * 32; i += 64) {
    unsigned w = gkeys[(size_t)b * S_ * 32 + i];
    int r = i >> 5, c2 = (i & 31) * 2;
    sK[r][c2] = __uint_as_float(w << 16);
    sK[r][c2 + 1] = __uint_as_float(w & 0xffff0000u);
  }
  __syncthreads();
  for (int h = 0; h < 3; ++h) {
    const float* lg = logits + ((size_t)b * NH_ + h) * S_;
    float lm = -3.4e38f;
    for (int s = t; s < S_; s += 64) lm = fmaxf(lm, lg[s]);
    red[t] = lm;
    __syncthreads();
    if (t == 0) {
      float m = red[0];
      for (int i = 1; i < 64; ++i) m = fmaxf(m, red[i]);
      sstat[0] = m;
    }
    __syncthreads();
    float gmax = sstat[0];
    float le = 0.f;
    for (int s = t; s < S_; s += 64) { float e = __expf(lg[s] - gmax); sp[s] = e; le += e; }
    red[t] = le;
    __syncthreads();
    if (t == 0) {
      float sum = 0.f;
      for (int i = 0; i < 64; ++i) sum += red[i];
      sstat[1] = sum;
    }
    __syncthreads();
    float inv = 1.f / sstat[1];
    float anym = (gmax > -1e8f) ? 1.f : 0.f;
    float acc = 0.f;
    for (int s = 0; s < S_; ++s) acc += sp[s] * sK[s][t];
    head_sum[((size_t)b * NH_ + h) * D_ + t] = acc * inv * anym;
    __syncthreads();
  }
}

// ======================================================================
// Kernel 6: dense_summary, context_enhanced, gate -> route weights
// ======================================================================
__global__ void __launch_bounds__(256) k_gate(
    const float* densef, const float* cand_sum, const float* ctx_sum,
    const float* comp_sum, const float* mask_cnt, const float* dense_w,
    const float* dense_b, const float* dense_g, const float* dense_be,
    const float* ctx_w, const float* ctx_b, const float* ctx_g,
    const float* ctx_be, const float* gate_w1, const float* gate_b1,
    const float* gate_g, const float* gate_be, const float* gate_w2,
    const float* gate_b2, float* comp_hist, float* dense_sum, float* ctx_enh,
    float* route_w) {
  __shared__ float sdf[DENSE_];
  __shared__ float sbuf[H_];
  __shared__ float sds[H_];
  __shared__ float sch[64], scs[64], sce[64], scand[64], sg[64];
  __shared__ float stats[2];
  int b = blockIdx.x, t = threadIdx.x;
  if (t < DENSE_) sdf[t] = densef[b * DENSE_ + t];
  if (t < 64) {
    float cnt = fmaxf(mask_cnt[b], 1.f);
    sch[t] = comp_sum[b * D_ + t] / cnt;
    scs[t] = ctx_sum[b * D_ + t];
    scand[t] = cand_sum[b * D_ + t];
  }
  __syncthreads();
  {
    float acc = dense_b[t];
    for (int k = 0; k < DENSE_; ++k) acc += sdf[k] * dense_w[k * H_ + t];
    sbuf[t] = acc;
  }
  __syncthreads();
  if (t == 0) {
    float m = 0.f; for (int i = 0; i < H_; ++i) m += sbuf[i]; m /= H_;
    float v = 0.f; for (int i = 0; i < H_; ++i) { float d0 = sbuf[i] - m; v += d0 * d0; } v /= H_;
    stats[0] = m; stats[1] = rsqrtf(v + 1e-5f);
  }
  __syncthreads();
  {
    float ds = siluf((sbuf[t] - stats[0]) * stats[1] * dense_g[t] + dense_be[t]);
    sds[t] = ds; dense_sum[(size_t)b * H_ + t] = ds;
  }
  __syncthreads();
  if (t < 64) {
    float acc = ctx_b[t];
    for (int k = 0; k < 64; ++k) acc += scs[k] * ctx_w[k * D_ + t];
    for (int k = 0; k < 64; ++k) acc += sch[k] * ctx_w[(64 + k) * D_ + t];
    sbuf[t] = acc;
  }
  __syncthreads();
  if (t == 0) {
    float m = 0.f; for (int i = 0; i < 64; ++i) m += sbuf[i]; m /= 64.f;
    float v = 0.f; for (int i = 0; i < 64; ++i) { float d0 = sbuf[i] - m; v += d0 * d0; } v /= 64.f;
    stats[0] = m; stats[1] = rsqrtf(v + 1e-5f);
  }
  __syncthreads();
  if (t < 64) {
    float ce = siluf((sbuf[t] - stats[0]) * stats[1] * ctx_g[t] + ctx_be[t]);
    sce[t] = ce; ctx_enh[(size_t)b * D_ + t] = ce; comp_hist[(size_t)b * D_ + t] = sch[t];
  }
  __syncthreads();
  if (t < 64) {
    float acc = gate_b1[t];
    for (int k = 0; k < 64; ++k) acc += scand[k] * gate_w1[k * D_ + t];
    for (int k = 0; k < 64; ++k) acc += sce[k] * gate_w1[(64 + k) * D_ + t];
    for (int k = 0; k < H_; ++k) acc += sds[k] * gate_w1[(128 + k) * D_ + t];
    sbuf[t] = acc;
  }
  __syncthreads();
  if (t == 0) {
    float m = 0.f; for (int i = 0; i < 64; ++i) m += sbuf[i]; m /= 64.f;
    float v = 0.f; for (int i = 0; i < 64; ++i) { float d0 = sbuf[i] - m; v += d0 * d0; } v /= 64.f;
    stats[0] = m; stats[1] = rsqrtf(v + 1e-5f);
  }
  __syncthreads();
  if (t < 64) sg[t] = siluf((sbuf[t] - stats[0]) * stats[1] * gate_g[t] + gate_be[t]);
  __syncthreads();
  if (t == 0) {
    float l0 = gate_b2[0], l1 = gate_b2[1], l2 = gate_b2[2];
    for (int k = 0; k < 64; ++k) {
      float g = sg[k];
      l0 += g * gate_w2[k * 3 + 0]; l1 += g * gate_w2[k * 3 + 1]; l2 += g * gate_w2[k * 3 + 2];
    }
    float mx = fmaxf(l0, fmaxf(l1, l2));
    float e0 = __expf(l0 - mx), e1 = __expf(l1 - mx), e2 = __expf(l2 - mx);
    float inv = 1.f / (e0 + e1 + e2);
    route_w[b * 3 + 0] = e0 * inv; route_w[b * 3 + 1] = e1 * inv; route_w[b * 3 + 2] = e2 * inv;
  }
}

// ======================================================================
// Kernel 7: fused feature concat + output MLP
// ======================================================================
__global__ void __launch_bounds__(256) k_out(
    const float* cand_sum, const float* ctx_sum, const float* ctx_enh,
    const float* head_sum, const float* comp_hist, const float* dense_sum,
    const float* route_w, const float* out_w1, const float* out_b1,
    const float* out_g, const float* out_be, const float* out_w2,
    const float* out_b2, const float* out_w3, const float* out_b3, float* out) {
  __shared__ float sf[11 * D_ + H_];
  __shared__ float sh1[H_];
  __shared__ float sh2[H_ / 2];
  __shared__ float stats[2];
  int b = blockIdx.x, t = threadIdx.x;
  if (t < 64) {
    float a = head_sum[((size_t)b * 3 + 0) * D_ + t];
    float c = head_sum[((size_t)b * 3 + 1) * D_ + t];
    float it = head_sum[((size_t)b * 3 + 2) * D_ + t];
    float r0 = route_w[b * 3 + 0], r1 = route_w[b * 3 + 1], r2 = route_w[b * 3 + 2];
    float g = a * r0 + c * r1 + it * r2;
    float mn = (a + c + it) * (1.f / 3.f);
    float v = ((a - mn) * (a - mn) + (c - mn) * (c - mn) + (it - mn) * (it - mn)) * (1.f / 3.f);
    float cd = cand_sum[b * D_ + t];
    sf[0 * 64 + t] = cd;
    sf[1 * 64 + t] = ctx_sum[b * D_ + t];
    sf[2 * 64 + t] = ctx_enh[b * D_ + t];
    sf[3 * 64 + t] = a;
    sf[4 * 64 + t] = c;
    sf[5 * 64 + t] = it;
    sf[6 * 64 + t] = g;
    sf[7 * 64 + t] = comp_hist[b * D_ + t];
    sf[8 * 64 + t] = cd * g;
    sf[9 * 64 + t] = fabsf(cd - g);
    sf[10 * 64 + t] = sqrtf(v);
  }
  sf[11 * 64 + t] = dense_sum[(size_t)b * H_ + t];
  __syncthreads();
  {
    float acc = out_b1[t];
    for (int k = 0; k < 11 * D_ + H_; ++k) acc += sf[k] * out_w1[k * H_ + t];
    sh1[t] = acc;
  }
  __syncthreads();
  if (t == 0) {
    float m = 0.f; for (int i = 0; i < H_; ++i) m += sh1[i]; m /= H_;
    float v = 0.f; for (int i = 0; i < H_; ++i) { float d0 = sh1[i] - m; v += d0 * d0; } v /= H_;
    stats[0] = m; stats[1] = rsqrtf(v + 1e-5f);
  }
  __syncthreads();
  float h1 = siluf((sh1[t] - stats[0]) * stats[1] * out_g[t] + out_be[t]);
  __syncthreads();
  sh1[t] = h1;
  __syncthreads();
  if (t < H_ / 2) {
    float acc = out_b2[t];
    for (int k = 0; k < H_; ++k) acc += sh1[k] * out_w2[k * (H_ / 2) + t];
    sh2[t] = siluf(acc);
  }
  __syncthreads();
  if (t == 0) {
    float acc = out_b3[0];
    for (int k = 0; k < H_ / 2; ++k) acc += sh2[k] * out_w3[k];
    out[b] = acc;
  }
}

// ======================================================================
// Host launcher
// ======================================================================
extern "C" void kernel_launch(void* const* d_in, const int* in_sizes, int n_in,
                              void* d_out, int out_size, void* d_ws, size_t ws_size,
                              hipStream_t stream) {
  (void)in_sizes; (void)n_in; (void)out_size; (void)ws_size;
  const int* cand_tok = (const int*)d_in[0];
  const int* ctx_tok = (const int*)d_in[1];
  const int* hist_tok = (const int*)d_in[2];
  const int* hist_pos = (const int*)d_in[3];
  const int* hist_grp = (const int*)d_in[4];
  const float* hist_time = (const float*)d_in[5];
  const unsigned char* cand_mask = (const unsigned char*)d_in[6];
  const unsigned char* ctx_mask = (const unsigned char*)d_in[7];
  const unsigned char* hist_mask = (const unsigned char*)d_in[8];
  const float* densef = (const float*)d_in[9];
  const float* a_b1 = (const float*)d_in[10];
  const float* a_b2 = (const float*)d_in[11];
  const float* a_w1 = (const float*)d_in[12];
  const float* a_w2 = (const float*)d_in[13];
  const float* c_b1 = (const float*)d_in[14];
  const float* c_b2 = (const float*)d_in[15];
  const float* c_w1 = (const float*)d_in[16];
  const float* c_w2 = (const float*)d_in[17];
  const float* i_b1 = (const float*)d_in[18];
  const float* i_b2 = (const float*)d_in[19];
  const float* i_w1 = (const float*)d_in[20];
  const float* i_w2 = (const float*)d_in[21];
  const float* comp_b = (const float*)d_in[22];
  const float* comp_be = (const float*)d_in[23];
  const float* comp_g = (const float*)d_in[24];
  const float* comp_w = (const float*)d_in[25];
  const float* ctx_b = (const float*)d_in[26];
  const float* ctx_be = (const float*)d_in[27];
  const float* ctx_g = (const float*)d_in[28];
  const float* ctx_w = (const float*)d_in[29];
  const float* dense_b = (const float*)d_in[30];
  const float* dense_be = (const float*)d_in[31];
  const float* dense_g = (const float*)d_in[32];
  const float* dense_w = (const float*)d_in[33];
  const float* gate_b1 = (const float*)d_in[34];
  const float* gate_b2 = (const float*)d_in[35];
  const float* gate_be = (const float*)d_in[36];
  const float* gate_g = (const float*)d_in[37];
  const float* gate_w1 = (const float*)d_in[38];
  const float* gate_w2 = (const float*)d_in[39];
  const float* grp_tab = (const float*)d_in[40];
  const float* out_b1 = (const float*)d_in[41];
  const float* out_b2 = (const float*)d_in[42];
  const float* out_b3 = (const float*)d_in[43];
  const float* out_be = (const float*)d_in[44];
  const float* out_g = (const float*)d_in[45];
  const float* out_w1 = (const float*)d_in[46];
  const float* out_w2 = (const float*)d_in[47];
  const float* out_w3 = (const float*)d_in[48];
  const float* pos_tab = (const float*)d_in[49];
  const float* time_b1 = (const float*)d_in[50];
  const float* time_b2 = (const float*)d_in[51];
  const float* time_w1 = (const float*)d_in[52];
  const float* time_w2 = (const float*)d_in[53];
  const float* tok_tab = (const float*)d_in[54];

  char* base = (char*)d_ws;
  size_t off = 0;
  auto alloc = [&](size_t bytes) {
    void* p = base + off;
    off = (off + bytes + 511) & ~(size_t)511;
    return p;
  };
  float* cand_sum = (float*)alloc(sizeof(float) * B_ * D_);
  float* ctx_sum = (float*)alloc(sizeof(float) * B_ * D_);
  float* comp_sum = (float*)alloc(sizeof(float) * B_ * D_);
  float* mask_cnt = (float*)alloc(sizeof(float) * B_);
  float* qw = (float*)alloc(sizeof(float) * B_ * NH_ * D_);
  float* logits = (float*)alloc(sizeof(float) * B_ * NH_ * S_);
  float* head_sum = (float*)alloc(sizeof(float) * B_ * NH_ * D_);
  float* ctx_enh = (float*)alloc(sizeof(float) * B_ * D_);
  float* dense_sum = (float*)alloc(sizeof(float) * B_ * H_);
  float* comp_hist = (float*)alloc(sizeof(float) * B_ * D_);
  float* route_w = (float*)alloc(sizeof(float) * B_ * 3);
  unsigned short* hist_emb =
      (unsigned short*)alloc(sizeof(unsigned short) * (size_t)B_ * S_ * D_);
  unsigned* tw2_sw = (unsigned*)alloc(sizeof(unsigned) * 2048);
  unsigned* cw_sw = (unsigned*)alloc(sizeof(unsigned) * 4096);
  unsigned* wq_sw = (unsigned*)alloc(sizeof(unsigned) * 3 * 2048);
  unsigned* wk_sw = (unsigned*)alloc(sizeof(unsigned) * 3 * 2048);
  unsigned* wc_sw = (unsigned*)alloc(sizeof(unsigned) * 3 * 2048);

  (void)hipMemsetAsync(comp_sum, 0, sizeof(float) * B_ * D_, stream);
  (void)hipMemsetAsync(mask_cnt, 0, sizeof(float) * B_, stream);

  k_prep<<<16, 256, 0, stream>>>(time_w2, comp_w, a_w1, c_w1, i_w1, tw2_sw, cw_sw,
                                 wq_sw, wk_sw, wc_sw);
  k_summ<<<B_, 64, 0, stream>>>(cand_tok, cand_mask, ctx_tok, ctx_mask, tok_tab,
                                cand_sum, ctx_sum);
  k_hist<<<B_ * NTILES, 32, 0, stream>>>(
      hist_time, hist_tok, hist_pos, hist_grp, hist_mask, tok_tab, pos_tab,
      grp_tab, time_w1, time_b1, time_b2, comp_b, comp_g, comp_be, tw2_sw, cw_sw,
      hist_emb, comp_sum, mask_cnt);
  k_qw<<<B_ / 16, 32, 0, stream>>>(cand_sum, wq_sw, a_b1, c_b1, i_b1, qw);
  k_din<<<B_ * NTILES, 32, 0, stream>>>(hist_emb, cand_sum, qw, wk_sw, wc_sw,
                                        a_w2, c_w2, i_w2, a_b2, c_b2, i_b2,
                                        hist_grp, hist_mask, logits);
  k_att<<<B_, 64, 0, stream>>>(logits, hist_emb, head_sum);
  k_gate<<<B_, 256, 0, stream>>>(densef, cand_sum, ctx_sum, comp_sum, mask_cnt,
                                 dense_w, dense_b, dense_g, dense_be, ctx_w,
                                 ctx_b, ctx_g, ctx_be, gate_w1, gate_b1, gate_g,
                                 gate_be, gate_w2, gate_b2, comp_hist, dense_sum,
                                 ctx_enh, route_w);
  k_out<<<B_, 256, 0, stream>>>(cand_sum, ctx_sum, ctx_enh, head_sum, comp_hist,
                                dense_sum, route_w, out_w1, out_b1, out_g, out_be,
                                out_w2, out_b2, out_w3, out_b3, (float*)d_out);
}
